// ChannelAttention_31078383353922
// MI455X (gfx1250) — matrix operand order repaired and checked by reading, not compiled, not run on hardware
//
#include <hip/hip_runtime.h>
#include <math.h>

// ---------------- problem constants ----------------
#define B_    32
#define P_    2048
#define NROWS (B_ * P_)      // 65536
#define C_    19
#define D_    512
#define H_    4
#define DH_   128
#define EPS_  1e-5f

// ---------------- tiling ----------------
#define MT    64             // rows per workgroup
#define TPB   512            // 16 waves
#define KC    32             // K chunk == WMMA K

typedef __attribute__((ext_vector_type(16))) _Float16 v16h;
typedef __attribute__((ext_vector_type(8)))  _Float16 h8;
typedef __attribute__((ext_vector_type(8)))  float    v8f;

union ABfrag { v16h v; h8 h[2]; };

// ------------------------------------------------------------------
// Kernel 1: convert Wq / Wo to f16 once (halves weight stream per WG)
// ------------------------------------------------------------------
__global__ void convert_w_kernel(const float* __restrict__ Wq,
                                 const float* __restrict__ Wo,
                                 _Float16* __restrict__ wqh,
                                 _Float16* __restrict__ woh) {
    int i = blockIdx.x * blockDim.x + threadIdx.x;
    if (i < D_ * D_) {
        wqh[i] = (_Float16)Wq[i];
        woh[i] = (_Float16)Wo[i];
    }
}

// ------------------------------------------------------------------
// Kernel 2: K/V depend only on channel embeddings: (19,512) each, fp32
// ------------------------------------------------------------------
__global__ void kv_kernel(const float* __restrict__ ce,
                          const float* __restrict__ Wk, const float* __restrict__ bk,
                          const float* __restrict__ Wv, const float* __restrict__ bv,
                          float* __restrict__ kbuf, float* __restrict__ vbuf) {
    int c = blockIdx.x;                       // 0..18
    __shared__ float ces[D_];
    for (int i = threadIdx.x; i < D_; i += blockDim.x)
        ces[i] = ce[(size_t)c * D_ + i];
    __syncthreads();
    for (int n = threadIdx.x; n < D_; n += blockDim.x) {
        const float* wkr = Wk + (size_t)n * D_;
        const float* wvr = Wv + (size_t)n * D_;
        float sk = 0.f, sv = 0.f;
        for (int d = 0; d < D_; d++) {
            float xv = ces[d];
            sk += xv * wkr[d];
            sv += xv * wvr[d];
        }
        kbuf[(size_t)c * D_ + n] = sk + bk[n];
        vbuf[(size_t)c * D_ + n] = sv + bv[n];
    }
}

// ------------------------------------------------------------------
// Kernel 3: fused q-GEMM + channel attention + o-GEMM + residual + LN
// Workgroup: 512 threads = 16 waves = 4 row-groups x 4 head-groups.
// Wave (r,h): rows 16r..16r+15, cols h*128..h*128+127 (8 WMMA tiles).
// ------------------------------------------------------------------
__global__ __launch_bounds__(TPB)
void fused_attn_kernel(const float* __restrict__ x,
                       const float* __restrict__ bq,
                       const float* __restrict__ bo,
                       const float* __restrict__ gamma,
                       const float* __restrict__ beta,
                       const float* __restrict__ kbuf,
                       const float* __restrict__ vbuf,
                       const _Float16* __restrict__ wqh,
                       const _Float16* __restrict__ woh,
                       float* __restrict__ out) {
    // ---- LDS ----
    __shared__ __align__(16) _Float16 a_chunk[MT * KC];     // 4 KB   A stage (f16)
    __shared__ __align__(16) _Float16 b_chunk[D_ * KC];     // 32 KB  B stage (f16, n-major)
    __shared__ __align__(16) float    ks_lds[C_ * D_];      // 38 KB  K
    __shared__ __align__(16) float    vs_lds[C_ * D_];      // 38 KB  V
    __shared__ __align__(16) float    sc_lds[MT * H_ * C_]; // 19 KB  scores/weights
    __shared__ __align__(16) float    stat_lds[MT * H_ * 2];// 2 KB   LN partials
    __shared__ __align__(16) float    mv_lds[MT * 2];       // mean / rstd

    const int tid   = threadIdx.x;
    const int lane  = tid & 31;
    const int wave  = tid >> 5;
    const int rgrp  = wave >> 2;          // 0..3
    const int cgrp  = wave & 3;           // 0..3  == head
    const int rbase = rgrp * 16;
    const int cbase = cgrp * DH_;
    const int hl    = lane & 15;
    const int half8 = (lane < 16) ? 0 : 8;
    const int koff8  = half8;             // A frag K offset
    const int koff16 = (lane < 16) ? 0 : 16; // unused; B frag uses koff8
    const int row0  = blockIdx.x * MT;

    // ---------------- GEMM1: q = x @ Wq^T ----------------
    v8f acc[8];
    #pragma unroll
    for (int t = 0; t < 8; t++)
        #pragma unroll
        for (int j = 0; j < 8; j++) acc[t][j] = 0.f;

    for (int kc0 = 0; kc0 < D_; kc0 += KC) {
        __syncthreads();
        {   // stage A: x (f32 -> f16), 64x32, one float4 per thread
            int r  = tid >> 3;
            int c4 = (tid & 7) * 4;
            float4 xv = *(const float4*)(x + (size_t)(row0 + r) * D_ + kc0 + c4);
            _Float16* dst = &a_chunk[r * KC + c4];
            dst[0] = (_Float16)xv.x; dst[1] = (_Float16)xv.y;
            dst[2] = (_Float16)xv.z; dst[3] = (_Float16)xv.w;
        }
        #pragma unroll
        for (int i = 0; i < 4; i++) {        // stage B: Wq f16, 512x32 n-major
            int idx = tid + i * TPB;         // 0..2047 groups of 8 halves
            int n   = idx >> 2;
            int c8  = (idx & 3) * 8;
            *(h8*)&b_chunk[n * KC + c8] =
                *(const h8*)(wqh + (size_t)n * D_ + kc0 + c8);
        }
        __syncthreads();

        ABfrag a;
        a.h[0] = *(const h8*)&a_chunk[(rbase + hl) * KC + koff8];
        a.h[1] = *(const h8*)&a_chunk[(rbase + hl) * KC + 16 + koff8];
        #pragma unroll
        for (int t = 0; t < 8; t++) {
            ABfrag b;
            const _Float16* bp = &b_chunk[(cbase + 16 * t + hl) * KC + koff8];
            b.h[0] = *(const h8*)bp;
            b.h[1] = *(const h8*)(bp + 16);
            acc[t] = __builtin_amdgcn_wmma_f32_16x16x32_f16(
                false, a.v, false, b.v, (short)0, acc[t], false, false);
        }
    }
    // + bq
    #pragma unroll
    for (int t = 0; t < 8; t++) {
        float bqv = bq[cbase + 16 * t + hl];
        #pragma unroll
        for (int j = 0; j < 8; j++) acc[t][j] += bqv;
    }

    // ---------------- load K/V into LDS ----------------
    __syncthreads();
    for (int i = tid; i < C_ * D_; i += TPB) {
        ks_lds[i] = kbuf[i];
        vs_lds[i] = vbuf[i];
    }
    __syncthreads();

    // ---------------- scores (fp32 VALU + shuffle reduce) ----------------
    const float sscale = 0.08838834764831845f;   // 1/sqrt(128)
    {
        const int h = cgrp;
        for (int c = 0; c < C_; c++) {
            float p[8];
            #pragma unroll
            for (int j = 0; j < 8; j++) p[j] = 0.f;
            #pragma unroll
            for (int t = 0; t < 8; t++) {
                float kv = ks_lds[c * D_ + h * DH_ + 16 * t + hl];
                #pragma unroll
                for (int j = 0; j < 8; j++) p[j] += acc[t][j] * kv;
            }
            #pragma unroll
            for (int j = 0; j < 8; j++) {
                float s = p[j];
                s += __shfl_xor(s, 1, 32);
                s += __shfl_xor(s, 2, 32);
                s += __shfl_xor(s, 4, 32);
                s += __shfl_xor(s, 8, 32);
                if (hl == 0) {
                    int m = rbase + j + half8;
                    sc_lds[(m * H_ + h) * C_ + c] = s * sscale;
                }
            }
        }
    }
    __syncthreads();

    // ---------------- softmax over C per (row, head) ----------------
    if (tid < MT * H_) {
        int m  = tid >> 2;
        int hh = tid & 3;
        float* s = &sc_lds[(m * H_ + hh) * C_];
        float mx = s[0];
        for (int c = 1; c < C_; c++) mx = fmaxf(mx, s[c]);
        float sum = 0.f;
        for (int c = 0; c < C_; c++) { float e = __expf(s[c] - mx); s[c] = e; sum += e; }
        float inv = 1.f / sum;
        for (int c = 0; c < C_; c++) s[c] *= inv;
    }
    __syncthreads();

    // ---------------- attended = weights @ V (overwrite acc) ----------------
    {
        const int h = cgrp;
        #pragma unroll
        for (int t = 0; t < 8; t++)
            #pragma unroll
            for (int j = 0; j < 8; j++) acc[t][j] = 0.f;
        for (int c = 0; c < C_; c++) {
            float wv[8];
            #pragma unroll
            for (int j = 0; j < 8; j++)
                wv[j] = sc_lds[((rbase + j + half8) * H_ + h) * C_ + c];
            #pragma unroll
            for (int t = 0; t < 8; t++) {
                float vv = vs_lds[c * D_ + h * DH_ + 16 * t + hl];
                #pragma unroll
                for (int j = 0; j < 8; j++) acc[t][j] += wv[j] * vv;
            }
        }
    }

    // ---------------- GEMM2: out = attended @ Wo^T ----------------
    v8f oacc[8];
    #pragma unroll
    for (int t = 0; t < 8; t++)
        #pragma unroll
        for (int j = 0; j < 8; j++) oacc[t][j] = 0.f;

    for (int kc0 = 0; kc0 < D_; kc0 += KC) {
        __syncthreads();
        {   // stage A from attended registers (owning head-waves), f32 -> f16
            int pcg = kc0 / DH_;                 // head that owns these K cols
            int t0  = (kc0 % DH_) / 16;          // first of 2 tiles
            if (cgrp == pcg) {
                #pragma unroll
                for (int tt = 0; tt < 2; tt++) {
                    #pragma unroll
                    for (int j = 0; j < 8; j++) {
                        int m = rbase + j + half8;
                        a_chunk[m * KC + 16 * tt + hl] = (_Float16)acc[t0 + tt][j];
                    }
                }
            }
        }
        #pragma unroll
        for (int i = 0; i < 4; i++) {            // stage B: Wo f16
            int idx = tid + i * TPB;
            int n   = idx >> 2;
            int c8  = (idx & 3) * 8;
            *(h8*)&b_chunk[n * KC + c8] =
                *(const h8*)(woh + (size_t)n * D_ + kc0 + c8);
        }
        __syncthreads();

        ABfrag a;
        a.h[0] = *(const h8*)&a_chunk[(rbase + hl) * KC + koff8];
        a.h[1] = *(const h8*)&a_chunk[(rbase + hl) * KC + 16 + koff8];
        #pragma unroll
        for (int t = 0; t < 8; t++) {
            ABfrag b;
            const _Float16* bp = &b_chunk[(cbase + 16 * t + hl) * KC + koff8];
            b.h[0] = *(const h8*)bp;
            b.h[1] = *(const h8*)(bp + 16);
            oacc[t] = __builtin_amdgcn_wmma_f32_16x16x32_f16(
                false, a.v, false, b.v, (short)0, oacc[t], false, false);
        }
    }

    // ---------------- + bo, residual h = x + out ----------------
    #pragma unroll
    for (int t = 0; t < 8; t++) {
        float bov = bo[cbase + 16 * t + hl];
        int n = cbase + 16 * t + hl;
        #pragma unroll
        for (int j = 0; j < 8; j++) {
            int m = rbase + j + half8;
            oacc[t][j] += bov + x[(size_t)(row0 + m) * D_ + n];   // x is L2-hot
        }
    }

    // ---------------- LayerNorm over D=512 ----------------
    #pragma unroll
    for (int j = 0; j < 8; j++) {
        float s = 0.f, ss = 0.f;
        #pragma unroll
        for (int t = 0; t < 8; t++) { float v = oacc[t][j]; s += v; ss += v * v; }
        s  += __shfl_xor(s, 1, 32);  ss += __shfl_xor(ss, 1, 32);
        s  += __shfl_xor(s, 2, 32);  ss += __shfl_xor(ss, 2, 32);
        s  += __shfl_xor(s, 4, 32);  ss += __shfl_xor(ss, 4, 32);
        s  += __shfl_xor(s, 8, 32);  ss += __shfl_xor(ss, 8, 32);
        if (hl == 0) {
            int m = rbase + j + half8;
            stat_lds[(m * 4 + cgrp) * 2 + 0] = s;
            stat_lds[(m * 4 + cgrp) * 2 + 1] = ss;
        }
    }
    __syncthreads();
    if (tid < MT) {
        float s = 0.f, ss = 0.f;
        #pragma unroll
        for (int g = 0; g < 4; g++) {
            s  += stat_lds[(tid * 4 + g) * 2 + 0];
            ss += stat_lds[(tid * 4 + g) * 2 + 1];
        }
        float mean = s * (1.f / (float)D_);
        float var  = ss * (1.f / (float)D_) - mean * mean;
        mv_lds[tid * 2 + 0] = mean;
        mv_lds[tid * 2 + 1] = rsqrtf(var + EPS_);
    }
    __syncthreads();

    #pragma unroll
    for (int t = 0; t < 8; t++) {
        int n = cbase + 16 * t + hl;
        float g  = gamma[n];
        float bt = beta[n];
        #pragma unroll
        for (int j = 0; j < 8; j++) {
            int m = rbase + j + half8;
            float mean = mv_lds[m * 2 + 0];
            float rstd = mv_lds[m * 2 + 1];
            out[(size_t)(row0 + m) * D_ + n] = (oacc[t][j] - mean) * rstd * g + bt;
        }
    }
}

// ------------------------------------------------------------------
extern "C" void kernel_launch(void* const* d_in, const int* in_sizes, int n_in,
                              void* d_out, int out_size, void* d_ws, size_t ws_size,
                              hipStream_t stream) {
    const float* x     = (const float*)d_in[0];
    const float* ce    = (const float*)d_in[1];
    const float* Wq    = (const float*)d_in[2];
    const float* bq    = (const float*)d_in[3];
    const float* Wk    = (const float*)d_in[4];
    const float* bk    = (const float*)d_in[5];
    const float* Wv    = (const float*)d_in[6];
    const float* bv    = (const float*)d_in[7];
    const float* Wo    = (const float*)d_in[8];
    const float* bo    = (const float*)d_in[9];
    const float* gamma = (const float*)d_in[10];
    const float* beta  = (const float*)d_in[11];
    float* out = (float*)d_out;

    // workspace layout (16B aligned offsets)
    char* ws = (char*)d_ws;
    float*    kbuf = (float*)(ws);                         // 19*512 f32 = 38912 B
    float*    vbuf = (float*)(ws + 38912);                 // 38912 B
    _Float16* wqh  = (_Float16*)(ws + 77824);              // 512*512 f16 = 524288 B
    _Float16* woh  = (_Float16*)(ws + 77824 + 524288);     // 524288 B

    convert_w_kernel<<<(D_ * D_ + 255) / 256, 256, 0, stream>>>(Wq, Wo, wqh, woh);
    kv_kernel<<<C_, 256, 0, stream>>>(ce, Wk, bk, Wv, bv, kbuf, vbuf);
    fused_attn_kernel<<<NROWS / MT, TPB, 0, stream>>>(
        x, bq, bo, gamma, beta, kbuf, vbuf, wqh, woh, out);
}
